// MambaBlock_84722524881200
// MI455X (gfx1250) — matrix operand order repaired and checked by reading, not compiled, not run on hardware
//
#include <hip/hip_runtime.h>
#include <math.h>

typedef __attribute__((ext_vector_type(16))) __bf16 v16bf;
typedef __attribute__((ext_vector_type(8)))  __bf16 v8bf;
typedef __attribute__((ext_vector_type(8)))  float  v8f;

constexpr int B  = 4;
constexpr int L  = 4096;
constexpr int DM = 64;    // d_model
constexpr int DI = 256;   // d_inner
constexpr int NS = 16;    // d_state
constexpr int XD = 48;    // padded x_dbl width (36 -> 48)

// ---------------------------------------------------------------------------
// WMMA helpers (CDNA5 16x16x32 bf16, wave32). Fragment packing follows ISA
// 7.12.2.  A (16x32): lane holds row M=lane%16; its 16 elements are two
// contiguous 8-element runs (k = kbase + hi*8 + 0..7 and +16).  B (32x16):
// lane holds col n=lane%16; its 16 elements are the same two runs
// (k = kbase + hi*8 + 0..7 and +16) when B is stored n-major in LDS.
// C/D f32: vgpr j -> M = j + 8*(lane/16), N = lane%16.
// ---------------------------------------------------------------------------
__device__ __forceinline__ v16bf load_a_frag(const __bf16* As, int lda, int kbase, int lane) {
  const int row = lane & 15, hi = lane >> 4;
  const __bf16* p = As + row * lda + kbase + hi * 8;
  v8bf a0 = *(const v8bf*)(p);        // k = kbase + hi*8 + 0..7
  v8bf a1 = *(const v8bf*)(p + 16);   // k = kbase + 16 + hi*8 + 0..7
  return __builtin_shufflevector(a0, a1, 0, 1, 2, 3, 4, 5, 6, 7,
                                 8, 9, 10, 11, 12, 13, 14, 15);
}

__device__ __forceinline__ v16bf load_b_frag(const __bf16* Bs /*[16][ldb]*/, int ldb,
                                             int kbase, int lane) {
  const int n = lane & 15, hi = lane >> 4;
  const __bf16* p = Bs + n * ldb + kbase + hi * 8;
  v8bf b0 = *(const v8bf*)(p);        // k = kbase + hi*8 + 0..7
  v8bf b1 = *(const v8bf*)(p + 16);   // k = kbase + 16 + hi*8 + 0..7
  return __builtin_shufflevector(b0, b1, 0, 1, 2, 3, 4, 5, 6, 7,
                                 8, 9, 10, 11, 12, 13, 14, 15);
}

__device__ __forceinline__ v8f wmma_bf16(v16bf a, v16bf b, v8f c) {
  // (neg_a, A, neg_b, B, c_mod, C, reuse_a, reuse_b)
  return __builtin_amdgcn_wmma_f32_16x16x32_bf16(false, a, false, b, (short)0, c, false, false);
}

__device__ __forceinline__ float sigmoidf_(float v) { return 1.0f / (1.0f + __expf(-v)); }
__device__ __forceinline__ float siluf_(float v)    { return v * sigmoidf_(v); }
__device__ __forceinline__ float softplusf_(float v){ return v > 20.0f ? v : log1pf(__expf(v)); }

// ---------------------------------------------------------------------------
// K1: xz[t, 0..511] = x[t_src, 0..63] @ in_proj^T     (per direction)
// grid (L/16, B), block 256 (8 waves). Each wave does 4 column tiles.
// B tiles are flat row-major copies of 16 contiguous weight rows.
// ---------------------------------------------------------------------------
__global__ void k_in_proj(const float* __restrict__ x, const float* __restrict__ W,
                          float* __restrict__ xz, int rev) {
  __shared__ alignas(16) __bf16 As[16 * 64];       // [token][k]
  __shared__ alignas(16) __bf16 Bs[8][16 * 64];    // per-wave [n][k]
  const int tile = blockIdx.x, b = blockIdx.y;
  const int tid = threadIdx.x, wave = tid >> 5, lane = tid & 31;

#pragma unroll
  for (int e = 0; e < 4; ++e) {                    // 1024 elements / 256 threads
    const int i = tid + e * 256;
    const int tl = i >> 6, k = i & 63;
    const int tg = tile * 16 + tl;
    const int ts = rev ? (L - 1 - tg) : tg;
    As[i] = (__bf16)x[((size_t)b * L + ts) * DM + k];
  }
  __syncthreads();

  for (int it = 0; it < 4; ++it) {
    const int ct = it * 8 + wave;                  // column tile 0..31
    for (int i = lane; i < 16 * 64; i += 32)       // flat copy of rows ct*16..+15
      Bs[wave][i] = (__bf16)W[ct * (16 * 64) + i];
    __syncthreads();

    v8f acc = {};
#pragma unroll
    for (int kc = 0; kc < 2; ++kc) {
      v16bf a = load_a_frag(As, 64, kc * 32, lane);
      v16bf w = load_b_frag(&Bs[wave][0], 64, kc * 32, lane);
      acc = wmma_bf16(a, w, acc);
    }
#pragma unroll
    for (int j = 0; j < 8; ++j) {
      const int m = j + ((lane >> 4) << 3);
      const int n = lane & 15;
      const int t = tile * 16 + m;
      xz[((size_t)b * L + t) * 512 + ct * 16 + n] = acc[j];
    }
    __syncthreads();
  }
}

// ---------------------------------------------------------------------------
// K2: causal depthwise conv(4) + bias + SiLU on x half of xz -> xc
// grid (L, B), block 256 (=DI)
// ---------------------------------------------------------------------------
__global__ void k_conv(const float* __restrict__ xz, const float* __restrict__ cw,
                       const float* __restrict__ cb, float* __restrict__ xc) {
  const int t = blockIdx.x, b = blockIdx.y, d = threadIdx.x;
  float v = cb[d];
#pragma unroll
  for (int j = 0; j < 4; ++j) {
    const int tt = t - 3 + j;
    if (tt >= 0) v += cw[d * 4 + j] * xz[((size_t)b * L + tt) * 512 + d];
  }
  xc[((size_t)b * L + t) * DI + d] = siluf_(v);
}

// ---------------------------------------------------------------------------
// K3a: x_dbl[t, 0..35] = xc @ x_proj^T  (padded to 48 cols)
// grid (L/16, B), block 128 (4 waves; waves 0..2 each own one column tile)
// ---------------------------------------------------------------------------
__global__ void k_xproj(const float* __restrict__ xc, const float* __restrict__ Wx,
                        float* __restrict__ xdbl) {
  __shared__ alignas(16) __bf16 As[16 * 256];      // [token][k]
  __shared__ alignas(16) __bf16 Bs[3][16 * 256];   // [n][k] per tile
  const int tile = blockIdx.x, b = blockIdx.y;
  const int tid = threadIdx.x, wave = tid >> 5, lane = tid & 31;

  for (int i = tid; i < 16 * 256; i += 128) {
    const int tl = i >> 8, k = i & 255;
    As[i] = (__bf16)xc[((size_t)b * L + tile * 16 + tl) * DI + k];
  }
  for (int i = tid; i < 3 * 16 * 256; i += 128) {  // rows 36..47 zero-padded
    const int o = i >> 8;                          // global output row 0..47
    Bs[0][i] = (__bf16)(o < 36 ? Wx[i] : 0.0f);
  }
  __syncthreads();

  if (wave < 3) {
    v8f acc = {};
#pragma unroll
    for (int kc = 0; kc < 8; ++kc) {
      v16bf a = load_a_frag(As, 256, kc * 32, lane);
      v16bf w = load_b_frag(&Bs[wave][0], 256, kc * 32, lane);
      acc = wmma_bf16(a, w, acc);
    }
#pragma unroll
    for (int j = 0; j < 8; ++j) {
      const int m = j + ((lane >> 4) << 3);
      const int n = lane & 15;
      const int t = tile * 16 + m;
      xdbl[((size_t)b * L + t) * XD + wave * 16 + n] = acc[j];
    }
  }
}

// ---------------------------------------------------------------------------
// K3b: dt = softplus(dt_raw @ dt_w^T + dt_b)   grid (L, B), block 256
// ---------------------------------------------------------------------------
__global__ void k_dt(const float* __restrict__ xdbl, const float* __restrict__ dtw,
                     const float* __restrict__ dtb, float* __restrict__ dt) {
  const int t = blockIdx.x, b = blockIdx.y, d = threadIdx.x;
  const float* xd = &xdbl[((size_t)b * L + t) * XD];
  float v = dtb[d];
#pragma unroll
  for (int r = 0; r < 4; ++r) v += xd[r] * dtw[d * 4 + r];
  dt[((size_t)b * L + t) * DI + d] = softplusf_(v);
}

// ---------------------------------------------------------------------------
// K4: selective scan. grid (B), block 256 (=DI). Thread d keeps h[16] in regs.
// dty holds dt on input and is overwritten in-place with y (+ x*D).
// ---------------------------------------------------------------------------
__global__ void k_scan(float* __restrict__ dty, const float* __restrict__ xc,
                       const float* __restrict__ xdbl, const float* __restrict__ Alog,
                       const float* __restrict__ Dv) {
  __shared__ float BC[32];
  const int b = blockIdx.x, d = threadIdx.x;
  float h[NS], Ae[NS];
#pragma unroll
  for (int n = 0; n < NS; ++n) { h[n] = 0.0f; Ae[n] = -__expf(Alog[d * NS + n]); }
  const float Dd = Dv[d];

  for (int t = 0; t < L; ++t) {
    if (threadIdx.x < 32)
      BC[threadIdx.x] = xdbl[((size_t)b * L + t) * XD + 4 + threadIdx.x];
    __syncthreads();

    const size_t idx = ((size_t)b * L + t) * DI + d;
    if (t + 8 < L) __builtin_prefetch(&dty[idx + 8 * DI], 0, 1);  // global_prefetch
    const float dtv = dty[idx];
    const float xv  = xc[idx];
    float y = 0.0f;
#pragma unroll
    for (int n = 0; n < NS; ++n) {
      h[n] = __expf(dtv * Ae[n]) * h[n] + dtv * BC[n] * xv;
      y += h[n] * BC[16 + n];
    }
    dty[idx] = y + xv * Dd;
    __syncthreads();
  }
}

// ---------------------------------------------------------------------------
// K5: gate + out_proj (WMMA) + residual + DyTanh + scatter to output.
// grid (L/16, B), block 128 (4 waves, one 16-col tile each).
// ---------------------------------------------------------------------------
__global__ void k_out(const float* __restrict__ y, const float* __restrict__ xz,
                      const float* __restrict__ x, const float* __restrict__ Wo,
                      const float* __restrict__ alphaP, const float* __restrict__ betaP,
                      const float* __restrict__ gammaP, const float* __restrict__ beta1P,
                      float* __restrict__ out, int rev) {
  __shared__ alignas(16) __bf16 As[16 * 256];      // [token][k]
  __shared__ alignas(16) __bf16 Bs[4][16 * 256];   // [n][k] per tile
  const int tile = blockIdx.x, b = blockIdx.y;
  const int tid = threadIdx.x, wave = tid >> 5, lane = tid & 31;

  for (int i = tid; i < 16 * 256; i += 128) {      // gated activations -> LDS
    const int tl = i >> 8, k = i & 255;
    const size_t idx = ((size_t)b * L + tile * 16 + tl);
    const float yv = y[idx * DI + k];
    const float zv = xz[idx * 512 + DI + k];
    As[i] = (__bf16)(yv * siluf_(zv));
  }
  for (int i = lane; i < 16 * 256; i += 32)        // flat copy of 16 weight rows
    Bs[wave][i] = (__bf16)Wo[wave * (16 * 256) + i];
  __syncthreads();

  v8f acc = {};
#pragma unroll
  for (int kc = 0; kc < 8; ++kc) {
    v16bf a = load_a_frag(As, 256, kc * 32, lane);
    v16bf w = load_b_frag(&Bs[wave][0], 256, kc * 32, lane);
    acc = wmma_bf16(a, w, acc);
  }

  const float al = alphaP[0], ga = gammaP[0];
#pragma unroll
  for (int j = 0; j < 8; ++j) {
    const int m = j + ((lane >> 4) << 3);
    const int n = lane & 15;
    const int o = wave * 16 + n;
    const int t = tile * 16 + m;
    const int to = rev ? (L - 1 - t) : t;          // un-flip backward dir
    const float v = acc[j] + x[((size_t)b * L + to) * DM + o];
    const float nv = ga * tanhf(al * v + beta1P[o]) + betaP[o];
    out[((size_t)b * L + to) * (2 * DM) + rev * DM + o] = nv;
  }
}

// ---------------------------------------------------------------------------
extern "C" void kernel_launch(void* const* d_in, const int* in_sizes, int n_in,
                              void* d_out, int out_size, void* d_ws, size_t ws_size,
                              hipStream_t stream) {
  (void)in_sizes; (void)n_in; (void)out_size; (void)ws_size;
  const float* x = (const float*)d_in[0];
  float* out = (float*)d_out;

  // workspace layout (floats)
  const size_t nBL = (size_t)B * L;
  float* ws   = (float*)d_ws;
  float* xz   = ws;                        // 2 * B*L*512
  float* xc   = xz   + 2 * nBL * 512;      // 2 * B*L*256
  float* xdbl = xc   + 2 * nBL * 256;      // 2 * B*L*48
  float* dty  = xdbl + 2 * nBL * XD;       // 2 * B*L*256  (dt, then y in-place)

  for (int dir = 0; dir < 2; ++dir) {
    const int pb = 1 + dir * 9;            // param base: pf at 1, pb at 10
    const float* Win  = (const float*)d_in[pb + 0];
    const float* cw   = (const float*)d_in[pb + 1];
    const float* cb   = (const float*)d_in[pb + 2];
    const float* Wx   = (const float*)d_in[pb + 3];
    const float* dtw  = (const float*)d_in[pb + 4];
    const float* dtb  = (const float*)d_in[pb + 5];
    const float* Alog = (const float*)d_in[pb + 6];
    const float* Dv   = (const float*)d_in[pb + 7];
    const float* Wo   = (const float*)d_in[pb + 8];
    const int nb = 19 + dir * 4;           // norm base: n1 at 19, n2 at 23
    const float* alpha = (const float*)d_in[nb + 0];
    const float* beta  = (const float*)d_in[nb + 1];
    const float* gamma = (const float*)d_in[nb + 2];
    const float* beta1 = (const float*)d_in[nb + 3];

    float* xz_d   = xz   + (size_t)dir * nBL * 512;
    float* xc_d   = xc   + (size_t)dir * nBL * 256;
    float* xdbl_d = xdbl + (size_t)dir * nBL * XD;
    float* dty_d  = dty  + (size_t)dir * nBL * 256;

    k_in_proj<<<dim3(L / 16, B), 256, 0, stream>>>(x, Win, xz_d, dir);
    k_conv   <<<dim3(L, B),      256, 0, stream>>>(xz_d, cw, cb, xc_d);
    k_xproj  <<<dim3(L / 16, B), 128, 0, stream>>>(xc_d, Wx, xdbl_d);
    k_dt     <<<dim3(L, B),      256, 0, stream>>>(xdbl_d, dtw, dtb, dty_d);
    k_scan   <<<dim3(B),         256, 0, stream>>>(dty_d, xc_d, xdbl_d, Alog, Dv);
    k_out    <<<dim3(L / 16, B), 128, 0, stream>>>(dty_d, xz_d, x, Wo,
                                                   alpha, beta, gamma, beta1, out, dir);
  }
}
